// EncoderSTB_27702539059234
// MI455X (gfx1250) — compile-verified, matrix operand order checked by reading, not run on hardware
//
#include <hip/hip_runtime.h>
#include <hip/hip_bf16.h>
#include <math.h>

// ---------------------------------------------------------------------------
// EncoderSTB for MI455X (gfx1250): wave32 + v_wmma_f32_16x16x32_f16 GEMMs,
// 256KB-LDS fused score/top-k/softmax/AV attention, O(1) sparse AV via
// colsum(v) identity, depthwise convs + 1x1 conv as a WMMA GEMM with
// async-to-LDS (ASYNCcnt) double-buffered A-tile staging.
// ---------------------------------------------------------------------------

typedef __attribute__((ext_vector_type(16))) _Float16 v16h;
typedef __attribute__((ext_vector_type(8)))  float    v8f;

#define NTOK   4096
#define CH     256
#define NHEAD  8
#define HDIM   32
#define HID    1024     // 4*C
#define K2     2048     // 2*HID
#define IMGW   64

// ---- workspace layout (bytes) ----
#define OFF_HF16    (size_t)0                       // [4096][256] f16    2 MB
#define OFF_QF16    (size_t)(2u<<20)                // [8][4096][32] f16  2 MB
#define OFF_KF16    (size_t)(4u<<20)                // [8][4096][32] f16  2 MB
#define OFF_VF32    (size_t)(6u<<20)                // [8][4096][32] f32  4 MB
#define OFF_CSUM    (size_t)(10u<<20)               // [8][32] f32
#define OFF_CTXH    (size_t)((10u<<20)+4096)        // [4096][256] f16    2 MB
#define OFF_X1      (size_t)((12u<<20)+4096)        // [4096][256] f32    4 MB
#define OFF_H2IMG   (size_t)((16u<<20)+4096)        // [256][4096] f32    4 MB
#define OFF_CATF16  (size_t)((20u<<20)+4096)        // [4096][2048] f16  16 MB
#define OFF_WQKVT   (size_t)((36u<<20)+4096)        // [768][256] f16
#define OFF_WPROJT  (size_t)((37u<<20)+4096)        // [256][256] f16
#define OFF_W1H     (size_t)((38u<<20)+4096)        // [256][2048] f16    1 MB

__device__ __forceinline__ int lane_id() { return threadIdx.x & 31; }

// Load a 16x32 f16 tile as a WMMA A operand in contiguous k order (also valid as the 32x16
// B-operand when the source is stored as Bt[n][k]):
//   lanes 0-15  : row = row0+lane,    k = k0+0..15   (16 contiguous halves)
//   lanes 16-31 : row = row0+lane-16, k = k0+16..31  (16 contiguous halves)
__device__ __forceinline__ v16h ld_tile(const _Float16* __restrict__ p,
                                        size_t row_stride, int row0, int k0) {
  int l = lane_id();
  const _Float16* q = p + (size_t)(row0 + (l & 15)) * row_stride
                        + (size_t)k0 + ((l >> 4) << 4);
  return *(const v16h*)q;
}

__device__ __forceinline__ v8f wmma_f16(v16h a, v16h b, v8f c) {
  return __builtin_amdgcn_wmma_f32_16x16x32_f16(false, a, false, b,
                                                (short)0, c, false, false);
}

// ---------------------------------------------------------------------------
// Weight conversion / transposition to f16 Bt[n][k] layouts
// ---------------------------------------------------------------------------
__global__ void cvt_weights_kernel(const float* __restrict__ qw,
                                   const float* __restrict__ kvw,
                                   const float* __restrict__ pw,
                                   const float* __restrict__ w1,
                                   _Float16* __restrict__ wqkvt,
                                   _Float16* __restrict__ wprojt,
                                   _Float16* __restrict__ w1h) {
  int i = blockIdx.x * blockDim.x + threadIdx.x;
  if (i < 768 * 256) {                       // Bt rows 0..255: q_w^T, 256..767: kv_w^T
    int n = i >> 8, k = i & 255;
    float v = (n < 256) ? qw[k * 256 + n] : kvw[k * 512 + (n - 256)];
    wqkvt[i] = (_Float16)v;
    return;
  }
  i -= 768 * 256;
  if (i < 256 * 256) {                       // proj_w^T
    int n = i >> 8, k = i & 255;
    wprojt[i] = (_Float16)pw[k * 256 + n];
    return;
  }
  i -= 256 * 256;
  if (i < 256 * 2048)                        // conv1_w already [c_out][c_in] = Bt
    w1h[i] = (_Float16)w1[i];
}

// ---------------------------------------------------------------------------
// LayerNorm (one 256-thread block per token row)
// ---------------------------------------------------------------------------
__global__ void ln1_kernel(const float* __restrict__ x, const float* __restrict__ g,
                           const float* __restrict__ b, _Float16* __restrict__ h) {
  __shared__ float red[CH];
  int row = blockIdx.x, t = threadIdx.x;
  float v = x[(size_t)row * CH + t];
  red[t] = v; __syncthreads();
  for (int s = 128; s > 0; s >>= 1) { if (t < s) red[t] += red[t + s]; __syncthreads(); }
  float mu = red[0] * (1.0f / CH); __syncthreads();
  float d = v - mu;
  red[t] = d * d; __syncthreads();
  for (int s = 128; s > 0; s >>= 1) { if (t < s) red[t] += red[t + s]; __syncthreads(); }
  float var = red[0] * (1.0f / CH);
  float y = d * rsqrtf(var + 1e-5f) * g[t] + b[t];
  h[(size_t)row * CH + t] = (_Float16)y;
}

__global__ void ln2_img_kernel(const float* __restrict__ x, const float* __restrict__ g,
                               const float* __restrict__ b, float* __restrict__ img) {
  __shared__ float red[CH];
  int row = blockIdx.x, t = threadIdx.x;
  float v = x[(size_t)row * CH + t];
  red[t] = v; __syncthreads();
  for (int s = 128; s > 0; s >>= 1) { if (t < s) red[t] += red[t + s]; __syncthreads(); }
  float mu = red[0] * (1.0f / CH); __syncthreads();
  float d = v - mu;
  red[t] = d * d; __syncthreads();
  for (int s = 128; s > 0; s >>= 1) { if (t < s) red[t] += red[t + s]; __syncthreads(); }
  float var = red[0] * (1.0f / CH);
  float y = d * rsqrtf(var + 1e-5f) * g[t] + b[t];
  img[(size_t)t * NTOK + row] = y;            // [channel][pixel] for the dwconvs
}

// ---------------------------------------------------------------------------
// QKV GEMM: [4096x256] @ Bt[768x256] -> q(f16,relu,*scale) / k(f16) / v(f32)
// per-head [h][n][d] layouts.  4 waves/block, one 16x16 WMMA tile per wave.
// ---------------------------------------------------------------------------
__global__ void gemm_qkv_kernel(const _Float16* __restrict__ h,
                                const _Float16* __restrict__ wt,
                                const float* __restrict__ qb,
                                const float* __restrict__ kvb,
                                _Float16* __restrict__ qf,
                                _Float16* __restrict__ kf,
                                float* __restrict__ vf) {
  int w  = threadIdx.x >> 5;
  int m0 = blockIdx.x * 16;
  int n0 = (blockIdx.y * 4 + w) * 16;
  v8f c = {};
  #pragma unroll
  for (int k0 = 0; k0 < CH; k0 += 32) {
    v16h a = ld_tile(h,  CH, m0, k0);
    v16h b = ld_tile(wt, CH, n0, k0);
    c = wmma_f16(a, b, c);
  }
  int l = lane_id();
  int n = n0 + (l & 15);
  int mb = m0 + ((l & 16) ? 8 : 0);
  const float SCALE = 0.17677669529663687f;   // D^-0.5, folded into q
  #pragma unroll
  for (int g = 0; g < 8; ++g) {
    int m = mb + g;
    float acc = c[g];
    if (n < 256) {
      float v = fmaxf(acc + qb[n], 0.0f) * SCALE;
      int head = n >> 5, d = n & 31;
      qf[((size_t)head * NTOK + m) * HDIM + d] = (_Float16)v;
    } else {
      int j = n - 256;                        // kv col: pair*256 + head*32 + d
      float v = acc + kvb[j];
      int head = (j >> 5) & 7, d = j & 31;
      if ((j >> 8) == 0) kf[((size_t)head * NTOK + m) * HDIM + d] = (_Float16)v;
      else               vf[((size_t)head * NTOK + m) * HDIM + d] = v;
    }
  }
}

// per-head column sums of v : colsum[h][d] = sum_n v[h][n][d]
__global__ void colsum_kernel(const float* __restrict__ vf, float* __restrict__ cs) {
  __shared__ float red[256];
  int head = blockIdx.x, t = threadIdx.x;
  int d = t & 31, chunk = t >> 5;             // 8 chunks of 512 rows
  float s = 0.0f;
  for (int n = chunk * 512; n < (chunk + 1) * 512; ++n)
    s += vf[((size_t)head * NTOK + n) * HDIM + d];
  red[t] = s; __syncthreads();
  for (int off = 128; off >= 32; off >>= 1) { if (t < off) red[t] += red[t + off]; __syncthreads(); }
  if (t < 32) cs[head * HDIM + d] = red[t];
}

// ---------------------------------------------------------------------------
// Fused attention: one block = (head, 16 query rows).
//  Phase 1: 4 waves sweep 256 column tiles, 1 WMMA each, scores -> 256KB LDS.
//  Phase 2: per-row exact top-8 (lane-local sorted list + wave32 tournament),
//           closed-form masked softmax, AV via colsum(v) + 8 gathers.
// ---------------------------------------------------------------------------
extern __shared__ float att_smem[];           // 16 * 4096 floats = 256 KB (WGP has 320 KB)

__global__ void attn_kernel(const _Float16* __restrict__ qf,
                            const _Float16* __restrict__ kf,
                            const float* __restrict__ vf,
                            const float* __restrict__ csum,
                            _Float16* __restrict__ ctxh) {
  int head = blockIdx.y;
  int rb   = blockIdx.x;                      // 16-row block
  int w    = threadIdx.x >> 5;
  int l    = lane_id();
  const _Float16* qh = qf + (size_t)head * NTOK * HDIM;
  const _Float16* kh = kf + (size_t)head * NTOK * HDIM;

  v16h a = ld_tile(qh, HDIM, rb * 16, 0);     // 16x32 query tile, reused 64x
  int rbase = (l & 16) ? 8 : 0;
  for (int ct = w; ct < NTOK / 16; ct += 4) {
    v16h b = ld_tile(kh, HDIM, ct * 16, 0);   // Bt rows are k rows: B[k][n]=K[n][k]
    v8f c = {};
    c = wmma_f16(a, b, c);                    // full K=D=32 in one instruction
    int col = ct * 16 + (l & 15);
    #pragma unroll
    for (int g = 0; g < 8; ++g) att_smem[(size_t)(rbase + g) * NTOK + col] = c[g];
  }
  __syncthreads();

  for (int rr = w; rr < 16; rr += 4) {        // each wave owns 4 rows
    const float* row = att_smem + (size_t)rr * NTOK;
    float lv[8]; int li[8];
    #pragma unroll
    for (int t = 0; t < 8; ++t) { lv[t] = -INFINITY; li[t] = -1; }
    for (int cidx = l; cidx < NTOK; cidx += 32) {   // lane-local top-8 (sorted)
      float s = row[cidx];
      if (s > lv[7]) {
        int t = 7;
        while (t > 0 && s > lv[t - 1]) { lv[t] = lv[t - 1]; li[t] = li[t - 1]; --t; }
        lv[t] = s; li[t] = cidx;
      }
    }
    // wave32 tournament: 8 rounds of butterfly arg-max over lane candidates
    float tv[8]; int ti[8]; int ptr = 0;
    #pragma unroll
    for (int t = 0; t < 8; ++t) {
      float cv = (ptr < 8) ? lv[ptr] : -INFINITY;
      int   ci = (ptr < 8) ? li[ptr] : -1;
      float myv = cv; int myi = ci;
      #pragma unroll
      for (int off = 16; off > 0; off >>= 1) {
        float ov = __shfl_xor(cv, off, 32);
        int   oi = __shfl_xor(ci, off, 32);
        if (ov > cv || (ov == cv && oi >= 0 && (ci < 0 || oi < ci))) { cv = ov; ci = oi; }
      }
      tv[t] = cv; ti[t] = ci;
      if (ptr < 8 && myi == ci && myv == cv) ++ptr;  // winner advances (indices unique)
    }
    // masked softmax in closed form: z_i = a_i on top-8, 0 elsewhere
    float m = fmaxf(tv[0], 0.0f);
    float e0 = __expf(-m);
    float denom = (float)(NTOK - 8) * e0;
    float p[8];
    #pragma unroll
    for (int t = 0; t < 8; ++t) { p[t] = __expf(tv[t] - m); denom += p[t]; }
    float inv  = 1.0f / denom;
    float base = e0 * inv;
    // AV: out[d] = base*colsum_v[d] + sum_t (p_t - base) * v[ti[t]][d]; lane=d
    float acc = base * csum[head * HDIM + l];
    #pragma unroll
    for (int t = 0; t < 8; ++t)
      acc += (p[t] * inv - base) * vf[((size_t)head * NTOK + ti[t]) * HDIM + l];
    int mrow = rb * 16 + rr;
    ctxh[(size_t)mrow * CH + head * HDIM + l] = (_Float16)acc;
  }
}

// ---------------------------------------------------------------------------
// Projection GEMM + residual: x1 = x + ctx @ proj_w + proj_b
// ---------------------------------------------------------------------------
__global__ void gemm_proj_kernel(const _Float16* __restrict__ ctx,
                                 const _Float16* __restrict__ wt,
                                 const float* __restrict__ pb,
                                 const float* __restrict__ x,
                                 float* __restrict__ x1) {
  int w  = threadIdx.x >> 5;
  int m0 = blockIdx.x * 16;
  int n0 = (blockIdx.y * 4 + w) * 16;
  v8f c = {};
  #pragma unroll
  for (int k0 = 0; k0 < CH; k0 += 32) {
    v16h a = ld_tile(ctx, CH, m0, k0);
    v16h b = ld_tile(wt,  CH, n0, k0);
    c = wmma_f16(a, b, c);
  }
  int l = lane_id();
  int n = n0 + (l & 15);
  int mb = m0 + ((l & 16) ? 8 : 0);
  #pragma unroll
  for (int g = 0; g < 8; ++g) {
    size_t o = (size_t)(mb + g) * CH + n;
    x1[o] = x[o] + c[g] + pb[n];
  }
}

// ---------------------------------------------------------------------------
// Depthwise 3x3 + 5x5 (channel multiplier 4) + ReLU, concat as f16 [pixel][2048]
// ---------------------------------------------------------------------------
__global__ void dwconv_kernel(const float* __restrict__ img,
                              const float* __restrict__ w3, const float* __restrict__ b3,
                              const float* __restrict__ w5, const float* __restrict__ b5,
                              _Float16* __restrict__ cat) {
  int idx = blockIdx.x * blockDim.x + threadIdx.x;      // n*2048 + oc
  int oc = idx & (K2 - 1);
  int n  = idx >> 11;
  int y = n >> 6, xx = n & 63;
  float acc;
  if (oc < HID) {
    int c = oc >> 2;                                    // groups=C, 4 outs per input ch
    const float* im = img + (size_t)c * NTOK;
    const float* w  = w3 + oc * 9;
    acc = b3[oc];
    #pragma unroll
    for (int dy = -1; dy <= 1; ++dy)
      #pragma unroll
      for (int dx = -1; dx <= 1; ++dx) {
        int yy = y + dy, xc = xx + dx;
        if ((unsigned)yy < 64u && (unsigned)xc < 64u)
          acc += w[(dy + 1) * 3 + (dx + 1)] * im[yy * IMGW + xc];
      }
  } else {
    int o = oc - HID;
    int c = o >> 2;
    const float* im = img + (size_t)c * NTOK;
    const float* w  = w5 + o * 25;
    acc = b5[o];
    #pragma unroll
    for (int dy = -2; dy <= 2; ++dy)
      #pragma unroll
      for (int dx = -2; dx <= 2; ++dx) {
        int yy = y + dy, xc = xx + dx;
        if ((unsigned)yy < 64u && (unsigned)xc < 64u)
          acc += w[(dy + 2) * 5 + (dx + 2)] * im[yy * IMGW + xc];
      }
  }
  cat[(size_t)n * K2 + oc] = (_Float16)fmaxf(acc, 0.0f);
}

// ---------------------------------------------------------------------------
// 1x1 conv as GEMM [4096x2048]@Bt[256x2048] + bias + residual -> d_out.
// The 16-row A (cat) tile is shared by all 4 waves of the block, so it is
// staged in LDS by the CDNA5 async copy engine (GLOBAL_LOAD_ASYNC_TO_LDS_B128,
// tracked by ASYNCcnt) with a 2-deep double buffer: waves 0-1 issue the
// per-lane 16B copies for k-step i+1 while all waves run WMMA on k-step i.
// Async loads complete in order, so after issuing the prefetch the issuing
// waves only need s_wait_asynccnt<=1 for the *current* buffer to be resident
// (<=0 on the final step); a workgroup barrier publishes it to waves 2-3.
// ---------------------------------------------------------------------------
__global__ void gemm_msfn_kernel(const _Float16* __restrict__ cat,
                                 const _Float16* __restrict__ w1,
                                 const float* __restrict__ b1,
                                 const float* __restrict__ x1,
                                 float* __restrict__ out) {
  __shared__ _Float16 atile[2][16 * 32];      // 2 x 1KB A-tile double buffer
  int tid = threadIdx.x;
  int w   = tid >> 5;
  int l   = lane_id();
  int m0  = blockIdx.x * 16;
  int n0  = (blockIdx.y * 4 + w) * 16;

  // per-lane async copy: threads 0..63 each move 16B (row = tid/4, 8-half seg)
  auto issue = [&](int bufi, int k0) {
    int row = tid >> 2, seg = tid & 3;
    unsigned lds = (unsigned)(uintptr_t)&atile[bufi][row * 32 + seg * 8];
    unsigned long long ga =
        (unsigned long long)(uintptr_t)(cat + (size_t)(m0 + row) * K2 + k0 + seg * 8);
    asm volatile("global_load_async_to_lds_b128 %0, %1, off"
                 :: "v"(lds), "v"(ga) : "memory");
  };

  if (tid < 64) issue(0, 0);
  v8f c = {};
  const int NSTEP = K2 / 32;
  for (int ks = 0; ks < NSTEP; ++ks) {
    int buf = ks & 1;
    bool more = (ks + 1 < NSTEP);
    if (tid < 64) {
      if (more) {
        issue(buf ^ 1, (ks + 1) * 32);
        asm volatile("s_wait_asynccnt 0x1" ::: "memory");  // current buf resident
      } else {
        asm volatile("s_wait_asynccnt 0x0" ::: "memory");  // last buf resident
      }
    }
    __syncthreads();
    v16h a = *(const v16h*)&atile[buf][(l & 15) * 32 + ((l >> 4) << 4)];
    v16h b = ld_tile(w1, K2, n0, ks * 32);
    c = wmma_f16(a, b, c);
    __syncthreads();                          // all waves done reading buf
  }

  int n  = n0 + (l & 15);
  int mb = m0 + ((l & 16) ? 8 : 0);
  #pragma unroll
  for (int g = 0; g < 8; ++g) {
    size_t o = (size_t)(mb + g) * CH + n;
    out[o] = x1[o] + c[g] + b1[n];
  }
}

// ---------------------------------------------------------------------------
extern "C" void kernel_launch(void* const* d_in, const int* in_sizes, int n_in,
                              void* d_out, int out_size, void* d_ws, size_t ws_size,
                              hipStream_t stream) {
  const float* x      = (const float*)d_in[0];
  const float* ln1_g  = (const float*)d_in[3];
  const float* ln1_b  = (const float*)d_in[4];
  const float* q_w    = (const float*)d_in[5];
  const float* q_b    = (const float*)d_in[6];
  const float* kv_w   = (const float*)d_in[7];
  const float* kv_b   = (const float*)d_in[8];
  const float* proj_w = (const float*)d_in[9];
  const float* proj_b = (const float*)d_in[10];
  const float* ln2_g  = (const float*)d_in[11];
  const float* ln2_b  = (const float*)d_in[12];
  const float* conv3w = (const float*)d_in[13];
  const float* conv3b = (const float*)d_in[14];
  const float* conv5w = (const float*)d_in[15];
  const float* conv5b = (const float*)d_in[16];
  const float* conv1w = (const float*)d_in[17];
  const float* conv1b = (const float*)d_in[18];

  char* ws = (char*)d_ws;
  _Float16* HF16   = (_Float16*)(ws + OFF_HF16);
  _Float16* QF16   = (_Float16*)(ws + OFF_QF16);
  _Float16* KF16   = (_Float16*)(ws + OFF_KF16);
  float*    VF32   = (float*)   (ws + OFF_VF32);
  float*    CSUM   = (float*)   (ws + OFF_CSUM);
  _Float16* CTXH   = (_Float16*)(ws + OFF_CTXH);
  float*    X1     = (float*)   (ws + OFF_X1);
  float*    H2IMG  = (float*)   (ws + OFF_H2IMG);
  _Float16* CATF16 = (_Float16*)(ws + OFF_CATF16);
  _Float16* WQKVT  = (_Float16*)(ws + OFF_WQKVT);
  _Float16* WPROJT = (_Float16*)(ws + OFF_WPROJT);
  _Float16* W1H    = (_Float16*)(ws + OFF_W1H);

  // attention block needs 256KB LDS (MI455X WGP: 320KB)
  (void)hipFuncSetAttribute((const void*)attn_kernel,
                            hipFuncAttributeMaxDynamicSharedMemorySize,
                            16 * NTOK * (int)sizeof(float));

  cvt_weights_kernel<<<(768*256 + 256*256 + 256*2048 + 255) / 256, 256, 0, stream>>>(
      q_w, kv_w, proj_w, conv1w, WQKVT, WPROJT, W1H);

  ln1_kernel<<<NTOK, CH, 0, stream>>>(x, ln1_g, ln1_b, HF16);

  gemm_qkv_kernel<<<dim3(NTOK / 16, 12), 128, 0, stream>>>(
      HF16, WQKVT, q_b, kv_b, QF16, KF16, VF32);

  colsum_kernel<<<NHEAD, 256, 0, stream>>>(VF32, CSUM);

  attn_kernel<<<dim3(NTOK / 16, NHEAD), 128, 16 * NTOK * sizeof(float), stream>>>(
      QF16, KF16, VF32, CSUM, CTXH);

  gemm_proj_kernel<<<dim3(NTOK / 16, 4), 128, 0, stream>>>(
      CTXH, WPROJT, proj_b, x, X1);

  ln2_img_kernel<<<NTOK, CH, 0, stream>>>(X1, ln2_g, ln2_b, H2IMG);

  dwconv_kernel<<<(NTOK * K2) / 256, 256, 0, stream>>>(
      H2IMG, conv3w, conv3b, conv5w, conv5b, CATF16);

  gemm_msfn_kernel<<<dim3(NTOK / 16, 4), 128, 0, stream>>>(
      CATF16, W1H, conv1b, X1, (float*)d_out);
}
